// DifferentiableRenderer_8581344658045
// MI455X (gfx1250) — matrix operand order repaired and checked by reading, not compiled, not run on hardware
//
#include <hip/hip_runtime.h>
#include <hip/hip_fp16.h>

typedef __attribute__((ext_vector_type(16))) _Float16 v16h;
typedef __attribute__((ext_vector_type(8)))  float    v8f;

#define IMG_H 256
#define IMG_W 256
#define IMG_HW (IMG_H * IMG_W)
#define NGAUSS 1024
#define NCHUNK (NGAUSS / 32)         // 32 chunks of 32 gaussians
#define NBATCH 2
#define C_FX 300.0f
#define C_FY 300.0f
#define C_CX 128.0f
#define C_CY 128.0f
#define C_EPS 1e-8f

#define THREADS_PER_BLOCK 128
#define WAVES_PER_BLOCK 4            // 128 / 32
#define TILES_PER_WAVE 8
#define TILES_PER_BATCH (IMG_HW / 16)                       // 4096
#define BLOCKS_PER_BATCH (TILES_PER_BATCH / (WAVES_PER_BLOCK * TILES_PER_WAVE)) // 128

// Fast hardware reciprocal (single v_rcp_f32, ~1 ulp) instead of the
// correctly-rounded div expansion.
__device__ __forceinline__ float fast_rcp(float x) {
    return __builtin_amdgcn_rcpf(x);
}

// ---------------------------------------------------------------------------
// Kernel 1: per-(batch, gaussian) camera transform + projection.
// Writes float4 {proj_x, proj_y, -0.5/var, opacity} into d_ws.
// ---------------------------------------------------------------------------
__global__ void splat_precompute_kernel(const float* __restrict__ positions,
                                        const float* __restrict__ opacities,
                                        const float* __restrict__ scales,
                                        const float* __restrict__ qvec,
                                        const float* __restrict__ tvec,
                                        float4* __restrict__ params) {
    int idx = blockIdx.x * blockDim.x + threadIdx.x;
    if (idx >= NBATCH * NGAUSS) return;
    int b = idx / NGAUSS;
    int n = idx - b * NGAUSS;

    float qw = qvec[b * 4 + 0];
    float qx = qvec[b * 4 + 1];
    float qy = qvec[b * 4 + 2];
    float qz = qvec[b * 4 + 3];
    float qinv = __builtin_amdgcn_rsqf(qw * qw + qx * qx + qy * qy + qz * qz);
    qw *= qinv; qx *= qinv; qy *= qinv; qz *= qinv;

    float r00 = 1.0f - 2.0f * (qy * qy + qz * qz);
    float r01 = 2.0f * (qx * qy - qz * qw);
    float r02 = 2.0f * (qx * qz + qy * qw);
    float r10 = 2.0f * (qx * qy + qz * qw);
    float r11 = 1.0f - 2.0f * (qx * qx + qz * qz);
    float r12 = 2.0f * (qy * qz - qx * qw);
    float r20 = 2.0f * (qx * qz - qy * qw);
    float r21 = 2.0f * (qy * qz + qx * qw);
    float r22 = 1.0f - 2.0f * (qx * qx + qy * qy);

    float px = positions[n * 3 + 0];
    float py = positions[n * 3 + 1];
    float pz = positions[n * 3 + 2];

    float cx = r00 * px + r01 * py + r02 * pz + tvec[b * 3 + 0];
    float cy = r10 * px + r11 * py + r12 * pz + tvec[b * 3 + 1];
    float cz = r20 * px + r21 * py + r22 * pz + tvec[b * 3 + 2];

    float invz = fast_rcp(cz);
    float ux = cx * invz * C_FX + C_CX;
    float uy = cy * invz * C_FY + C_CY;

    float s = scales[n];
    float ninv2var = -0.5f * fast_rcp(s * s);

    params[idx] = make_float4(ux, uy, ninv2var, opacities[n]);
}

// ---------------------------------------------------------------------------
// Kernel 2: tiled renderer.
// Per 32-gaussian chunk, each wave builds
//   A (16 pix x 32 gauss, f16)  -- weights computed per lane (exp stream)
//   B (32 gauss x 16 cols, f16) -- cols 0..2 = RGB, col 3 = 1, rest 0,
//                                  PRE-PACKED in LDS as per-lane v16h rows
// and accumulates D = A*B + D with v_wmma_f32_16x16x32_f16:
// D columns 0..2 = per-pixel numerators, column 3 = den.
// ---------------------------------------------------------------------------
__global__ void __launch_bounds__(THREADS_PER_BLOCK)
splat_render_kernel(const float4* __restrict__ params,
                    const float* __restrict__ colors,
                    float* __restrict__ out) {
    __shared__ float4 sp[NGAUSS];                          // 16 KB params
    __shared__ __align__(32) _Float16 sb[NCHUNK][32][16];  // 32 KB B fragments

    const int b = blockIdx.x / BLOCKS_PER_BATCH;
    const int blockInBatch = blockIdx.x - b * BLOCKS_PER_BATCH;

    // Cooperative load: params -> LDS
    for (int n = threadIdx.x; n < NGAUSS; n += THREADS_PER_BLOCK) {
        sp[n] = params[b * NGAUSS + n];
    }
    // Cooperative build: B fragments in exact per-lane WMMA layout.
    // Item = (chunk c, lane ln): col = ln&15, K-base = (ln>>4)*16,
    // element e -> gaussian c*32 + Kbase + e, channel col.
    for (int item = threadIdx.x; item < NCHUNK * 32; item += THREADS_PER_BLOCK) {
        const int c   = item >> 5;
        const int ln  = item & 31;
        const int col = ln & 15;
        const int gb  = c * 32 + (ln >> 4) * 16;
        v16h frag;
#pragma unroll
        for (int e = 0; e < 16; ++e) {
            float v = 0.0f;
            if (col < 3)       v = colors[(gb + e) * 3 + col];
            else if (col == 3) v = 1.0f;
            frag[e] = (_Float16)v;
        }
        *(v16h*)&sb[c][ln][0] = frag;
    }
    __syncthreads();

    const int lane  = threadIdx.x & 31;
    const int wave  = threadIdx.x >> 5;
    const int col   = lane & 15;   // A row / D column index
    const int hi    = lane >> 4;   // which half-wave
    const int aKoff = hi * 8;      // A-matrix K base for this half is aKoff*2 (0 or 16)

    const int waveGlobal = blockInBatch * WAVES_PER_BLOCK + wave;

    for (int t = 0; t < TILES_PER_WAVE; ++t) {
        const int tile    = waveGlobal * TILES_PER_WAVE + t;
        const int pixBase = tile * 16;
        const int mypix   = pixBase + col;          // pixel this lane's A-row covers
        const float pxf   = (float)(mypix & (IMG_W - 1));
        const float pyf   = (float)(mypix >> 8);

        v8f acc = {};

#pragma unroll 2
        for (int c = 0; c < NCHUNK; ++c) {
            const int g0 = c * 32;
            // B: one aligned 32B LDS read (2x ds_load_b128), no divergence.
            const v16h bv = *(const v16h*)&sb[c][lane][0];

            v16h a;
#pragma unroll
            for (int e = 0; e < 16; ++e) {
                // A element e in this lane maps to K = hi*16 + e
                const int ka = aKoff * 2 + e;
                const float4 p = sp[g0 + ka];
                const float dx = p.x - pxf;
                const float dy = p.y - pyf;
                const float wv = p.w * __expf((dx * dx + dy * dy) * p.z);
                a[e] = (_Float16)wv;
            }
            // D(16x16 f32) += A(16x32 f16) x B(32x16 f16)
            acc = __builtin_amdgcn_wmma_f32_16x16x32_f16(
                false, a, false, bv, (short)0, acc, false, false);
        }

        // den lives in D column 3 -> broadcast within each half-wave.
        const int srcLane = (lane & 16) | 3;
#pragma unroll
        for (int j = 0; j < 8; ++j) {
            const float den = __shfl(acc[j], srcLane, 32);
            acc[j] = acc[j] * fast_rcp(den + C_EPS);
        }

        if (col < 3) {
#pragma unroll
            for (int j = 0; j < 8; ++j) {
                const int row = j + hi * 8;              // D row = pixel in tile
                out[(b * 3 + col) * IMG_HW + pixBase + row] = acc[j];
            }
        }
    }
}

extern "C" void kernel_launch(void* const* d_in, const int* in_sizes, int n_in,
                              void* d_out, int out_size, void* d_ws, size_t ws_size,
                              hipStream_t stream) {
    (void)in_sizes; (void)n_in; (void)out_size; (void)ws_size;
    const float* positions = (const float*)d_in[0];
    const float* colors    = (const float*)d_in[1];
    const float* opacities = (const float*)d_in[2];
    const float* scales    = (const float*)d_in[3];
    const float* qvec      = (const float*)d_in[4];
    const float* tvec      = (const float*)d_in[5];
    // d_in[6] = pixel_coords: implicit (pixel grid is regular), unused.

    float4* params = (float4*)d_ws;   // NBATCH*NGAUSS float4 = 32 KB

    splat_precompute_kernel<<<(NBATCH * NGAUSS + 127) / 128, 128, 0, stream>>>(
        positions, opacities, scales, qvec, tvec, params);

    splat_render_kernel<<<NBATCH * BLOCKS_PER_BATCH, THREADS_PER_BLOCK, 0, stream>>>(
        params, colors, (float*)d_out);
}
